// NeuralSDE_28664611733667
// MI455X (gfx1250) — matrix operand order repaired and checked by reading, not compiled, not run on hardware
//
#include <hip/hip_runtime.h>
#include <math.h>

typedef __attribute__((ext_vector_type(16))) _Float16 v16h;
typedef __attribute__((ext_vector_type(8)))  _Float16 v8h;
typedef __attribute__((ext_vector_type(8)))  float    v8f;

#define HID    64
#define NOISE  16
#define DATA   8
#define WIDTH  128
#define TPTS   1025
#define NSTEP  1024
#define BATCH  512
#define BM     16      // batch tile per block (WMMA M)
#define K0P    96      // layer-0 K padded: 65 -> 96 (3 x K32 fragments)
#define NT     512     // threads per block (16 waves, 4 per SIMD32)

// ---------------- fast math on gfx1250 hardware units ----------------------
__device__ __forceinline__ float fast_rcp(float x) {
#if __has_builtin(__builtin_amdgcn_rcpf)
    return __builtin_amdgcn_rcpf(x);
#else
    return 1.0f / x;
#endif
}
__device__ __forceinline__ float fast_tanh(float x) {
#if __has_builtin(__builtin_amdgcn_tanhf)
    return __builtin_amdgcn_tanhf(x);      // v_tanh_f32
#else
    float e = __expf(2.0f * x);
    return (e - 1.0f) * fast_rcp(e + 1.0f);
#endif
}
__device__ __forceinline__ float lipswish(float x) {
    return 0.909f * x * fast_rcp(1.0f + __expf(-x));   // v_exp + v_rcp
}

// ---------------- async global->LDS (CDNA5), guarded ----------------------
#if __has_builtin(__builtin_amdgcn_global_load_async_to_lds_b32) && \
    __has_builtin(__builtin_amdgcn_s_wait_asynccnt)
#define HAVE_ASYNC_LDS 1
#else
#define HAVE_ASYNC_LDS 0
#endif

typedef __attribute__((address_space(1))) int gbl_i32_t;
typedef __attribute__((address_space(3))) int lds_i32_t;

__device__ __forceinline__ void async_load_f32(const float* g, float* l) {
#if HAVE_ASYNC_LDS
    // generic->AS via inttoptr; generic LDS addr low 32 bits == LDS offset
    __builtin_amdgcn_global_load_async_to_lds_b32(
        (gbl_i32_t*)(unsigned long long)g,
        (lds_i32_t*)(unsigned int)(unsigned long long)l, 0, 0);
#else
    *l = *g;
#endif
}
__device__ __forceinline__ void wait_async() {
#if HAVE_ASYNC_LDS
    __builtin_amdgcn_s_wait_asynccnt(0);
#endif
}

// ---------------------------------------------------------------------------
// fp32 -> f16 weight conversion with K padding. dst[n*Kdst+k] row-major.
// ---------------------------------------------------------------------------
__global__ void convert_pad_f16(const float* __restrict__ src,
                                _Float16* __restrict__ dst,
                                int N, int Ksrc, int Kdst) {
    int total = N * Kdst;
    for (int i = blockIdx.x * blockDim.x + threadIdx.x; i < total;
         i += gridDim.x * blockDim.x) {
        int n = i / Kdst, k = i - n * Kdst;
        dst[i] = (_Float16)(k < Ksrc ? src[n * Ksrc + k] : 0.0f);
    }
}

// ---------------------------------------------------------------------------
// One-shot initial MLP: y0 = MLP_relu(z0), plus readout row t=0.
// ---------------------------------------------------------------------------
__global__ void sde_init(const float* __restrict__ z0,
                         const float* __restrict__ iW0, const float* __restrict__ ib0,
                         const float* __restrict__ iW1, const float* __restrict__ ib1,
                         const float* __restrict__ iW2, const float* __restrict__ ib2,
                         const float* __restrict__ roW, const float* __restrict__ rob,
                         float* __restrict__ yw, float* __restrict__ out) {
    __shared__ float z[8], h0[WIDTH], h1[WIDTH], y0s[HID];
    int b = blockIdx.x, j = threadIdx.x;
    if (j < 8) z[j] = z0[b * 8 + j];
    __syncthreads();
    float a = ib0[j];
    for (int k = 0; k < 8; ++k) a += z[k] * iW0[j * 8 + k];
    h0[j] = fmaxf(a, 0.0f);
    __syncthreads();
    a = ib1[j];
    for (int k = 0; k < WIDTH; ++k) a += h0[k] * iW1[j * WIDTH + k];
    h1[j] = fmaxf(a, 0.0f);
    __syncthreads();
    if (j < HID) {
        a = ib2[j];
        for (int k = 0; k < WIDTH; ++k) a += h1[k] * iW2[j * WIDTH + k];
        y0s[j] = a;
        yw[b * HID + j] = a;
    }
    __syncthreads();
    if (j < DATA) {
        a = rob[j];
        for (int h = 0; h < HID; ++h) a += y0s[h] * roW[j * HID + h];
        out[(size_t)b * TPTS * DATA + j] = a;
    }
}

// ---------------------------------------------------------------------------
// WMMA fragment helpers (CDNA5 16-bit A 16x32 / B 32x16 layouts).
// ---------------------------------------------------------------------------
__device__ __forceinline__ void load_afrag(const _Float16* __restrict__ X, int ld,
                                           int kfrags, v16h* a) {
    const int lane = threadIdx.x & 31;
    const int row = lane & 15, hi = lane >> 4;
#pragma unroll
    for (int kf = 0; kf < kfrags; ++kf) {
        const int kb = kf * 32;
        v8h lo = *(const v8h*)(X + row * ld + kb + hi * 8);
        v8h hi8 = *(const v8h*)(X + row * ld + kb + 16 + hi * 8);
        a[kf] = __builtin_shufflevector(lo, hi8, 0, 1, 2, 3, 4, 5, 6, 7,
                                        8, 9, 10, 11, 12, 13, 14, 15);
    }
}

__device__ __forceinline__ v8f wmma_bfrags(const v16h* a,
                                           const _Float16* __restrict__ W,
                                           int ld, int nbase, int kfrags) {
    const int lane = threadIdx.x & 31;
    const int row = lane & 15, hi = lane >> 4;
    v8f acc = {};
#pragma unroll
    for (int kf = 0; kf < kfrags; ++kf) {
        const _Float16* wp = W + (size_t)(nbase + row) * ld + kf * 32 + hi * 8;
        v8h blo = *(const v8h*)(wp);
        v8h bhi = *(const v8h*)(wp + 16);
        v16h b = __builtin_shufflevector(blo, bhi, 0, 1, 2, 3, 4, 5, 6, 7,
                                         8, 9, 10, 11, 12, 13, 14, 15);
        acc = __builtin_amdgcn_wmma_f32_16x16x32_f16(false, a[kf], false, b,
                                                     (short)0, acc, false, false);
    }
    return acc;
}

// lipswish epilogue -> f16 activations (C layout: VGPR r -> M=r+8*hi, N=col)
__device__ __forceinline__ void store_lipswish(v8f acc, const float* __restrict__ bias,
                                               int nbase, _Float16* __restrict__ Y) {
    const int lane = threadIdx.x & 31;
    const int hi = lane >> 4, col = lane & 15;
    const int n = nbase + col;
    const float bv = bias[n];
#pragma unroll
    for (int r = 0; r < 8; ++r) {
        const int m = r + 8 * hi;
        Y[m * WIDTH + n] = (_Float16)lipswish(acc[r] + bv);
    }
}

// diffusion epilogue: tanh * (cscale*sqrt(dt)), times raw dW, 16-lane reduce
__device__ __forceinline__ void diff_epilogue(v8f acc, int tile,
                                              const float* __restrict__ cb2,
                                              const float* __restrict__ csc,
                                              const float* __restrict__ dwc,
                                              float* __restrict__ g) {
    const int lane = threadIdx.x & 31;
    const int hi = lane >> 4, col = lane & 15;
    const int n = tile * 16 + col;
    const float bv = cb2[n], cs = csc[n];
#pragma unroll
    for (int r = 0; r < 8; ++r) {
        const int m = r + 8 * hi;
        float v = fast_tanh(acc[r] + bv) * cs * dwc[m * NOISE + col];
        v += __shfl_xor(v, 1, 32);
        v += __shfl_xor(v, 2, 32);
        v += __shfl_xor(v, 4, 32);
        v += __shfl_xor(v, 8, 32);
        if (col == 0) g[m * HID + tile] = v;
    }
}

__device__ __forceinline__ void drift_epilogue(v8f acc, int tile,
                                               const float* __restrict__ vb2,
                                               const float* __restrict__ vsc,
                                               float dt, float* __restrict__ f) {
    const int lane = threadIdx.x & 31;
    const int hi = lane >> 4, col = lane & 15;
    const int n = tile * 16 + col;
    const float bv = vb2[n], vs = vsc[n];
#pragma unroll
    for (int r = 0; r < 8; ++r) {
        const int m = r + 8 * hi;
        f[m * HID + n] = fast_tanh(acc[r] + bv) * vs * dt;
    }
}

// ---------------------------------------------------------------------------
// Persistent scan: 1 block = 16 samples x 1024 Euler-Heun steps, 16 waves.
// LDS-resident: both branches' layer-0/1 weights + vW2 (f16, 128KB),
// biases/scales/readout (~12KB), activations/state (~37KB).
// cW2 (1024x128 f16) streamed from global (L2-resident).
// dW tile double-buffered via async global->LDS, issued one step ahead.
// ---------------------------------------------------------------------------
__global__ void __launch_bounds__(NT)
sde_scan(const float* __restrict__ ts, const float* __restrict__ dW,
         const _Float16* __restrict__ gWlds,   // [vW0h|vW1h|vW2h|cW0h|cW1h]
         const _Float16* __restrict__ gW2c,    // cW2h [1024][128]
         const float* __restrict__ vb0, const float* __restrict__ vb1,
         const float* __restrict__ vb2, const float* __restrict__ vscale,
         const float* __restrict__ cb0, const float* __restrict__ cb1,
         const float* __restrict__ cb2, const float* __restrict__ cscale,
         const float* __restrict__ roW, const float* __restrict__ rob,
         const float* __restrict__ yw, float* __restrict__ out) {
    extern __shared__ __align__(32) char smem[];
    _Float16* wv0 = (_Float16*)smem;                 // 128*96
    _Float16* wv1 = wv0 + WIDTH * K0P;               // 128*128
    _Float16* wv2 = wv1 + WIDTH * WIDTH;             // 64*128
    _Float16* wc0 = wv2 + HID * WIDTH;               // 128*96
    _Float16* wc1 = wc0 + WIDTH * K0P;               // 128*128
    _Float16* xh  = wc1 + WIDTH * WIDTH;             // 16*96
    _Float16* h0v = xh + BM * K0P;                   // 16*128
    _Float16* h0c = h0v + BM * WIDTH;                // 16*128
    _Float16* h1v = h0c + BM * WIDTH;                // 16*128
    _Float16* h1c = h1v + BM * WIDTH;                // 16*128
    float* ys   = (float*)(h1c + BM * WIDTH);        // 16*64
    float* f0   = ys + BM * HID;
    float* g0   = f0 + BM * HID;
    float* g1   = g0 + BM * HID;
    float* dwb0 = g1 + BM * HID;                     // 16*16 (raw dW, buf 0)
    float* dwb1 = dwb0 + BM * NOISE;                 // 16*16 (raw dW, buf 1)
    float* vb0s = dwb1 + BM * NOISE;                 // 128
    float* vb1s = vb0s + WIDTH;                      // 128
    float* vb2s = vb1s + WIDTH;                      // 64
    float* vscs = vb2s + HID;                        // 64
    float* cb0s = vscs + HID;                        // 128
    float* cb1s = cb0s + WIDTH;                      // 128
    float* cb2s = cb1s + WIDTH;                      // 1024
    float* cscs = cb2s + HID * NOISE;                // 1024 (cscale * sqrt(dt))
    float* roWs = cscs + HID * NOISE;                // 8*64
    float* robs = roWs + DATA * HID;                 // 8

    const int tid  = threadIdx.x;
    const int wave = tid >> 5;
    const int base = blockIdx.x * BM;

    const float dt  = ts[1] - ts[0];
    const float sdt = sqrtf(dt);

    // stage LDS-resident weights (131072 B) as 16B chunks
    {
        const uint4* src = (const uint4*)gWlds;
        uint4* dst = (uint4*)smem;
        const int nvec = (2 * WIDTH * K0P + 2 * WIDTH * WIDTH + HID * WIDTH) * 2 / 16;
        for (int i = tid; i < nvec; i += NT) dst[i] = src[i];
    }
    // stage biases / scales / readout (sqrt(dt) folded into cscale)
    for (int i = tid; i < WIDTH; i += NT) { vb0s[i] = vb0[i]; vb1s[i] = vb1[i];
                                            cb0s[i] = cb0[i]; cb1s[i] = cb1[i]; }
    for (int i = tid; i < HID; i += NT)   { vb2s[i] = vb2[i]; vscs[i] = vscale[i]; }
    for (int i = tid; i < HID * NOISE; i += NT) { cb2s[i] = cb2[i];
                                                  cscs[i] = cscale[i] * sdt; }
    for (int i = tid; i < DATA * HID; i += NT) roWs[i] = roW[i];
    for (int i = tid; i < DATA; i += NT) robs[i] = rob[i];
    for (int i = tid; i < BM * K0P; i += NT) xh[i] = (_Float16)0;   // K padding
    for (int i = tid; i < BM * HID; i += NT) ys[i] = yw[base * HID + i];
    // build x = [t0, y0] and kick off async dW(0)
    for (int i = tid; i < BM * (HID + 1); i += NT) {
        int m = i / (HID + 1), k = i - m * (HID + 1);
        xh[m * K0P + k] = (_Float16)(k == 0 ? ts[0] : ys[m * HID + (k - 1)]);
    }
    if (tid < BM * NOISE)
        async_load_f32(dW + (size_t)(base + (tid >> 4)) * NOISE + (tid & 15),
                       dwb0 + tid);

    for (int st = 0; st < NSTEP; ++st) {
        float* dwc = (st & 1) ? dwb1 : dwb0;
        float* dwn = (st & 1) ? dwb0 : dwb1;
        wait_async();                  // dW(st) landed (issued one step ago)
        __syncthreads();               // xh(st), dwc visible

        // ---- fused layer 0 (drift + diffusion), 16 tiles over 16 waves ----
        {
            v16h ax[3];
            load_afrag(xh, K0P, 3, ax);                       // shared A
            const _Float16* W = (wave < 8) ? wv0 : wc0;
            const float* bs   = (wave < 8) ? vb0s : cb0s;
            _Float16* dst     = (wave < 8) ? h0v : h0c;
            const int nb = (wave & 7) * 16;
            v8f acc = wmma_bfrags(ax, W, K0P, nb, 3);
            store_lipswish(acc, bs, nb, dst);
        }
        __syncthreads();

        // ---- fused layer 1, 16 tiles over 16 waves ----
        {
            const _Float16* Xs = (wave < 8) ? h0v : h0c;
            const _Float16* W  = (wave < 8) ? wv1 : wc1;
            const float* bs    = (wave < 8) ? vb1s : cb1s;
            _Float16* dst      = (wave < 8) ? h1v : h1c;
            v16h a1[4];
            load_afrag(Xs, WIDTH, 4, a1);
            const int nb = (wave & 7) * 16;
            v8f acc = wmma_bfrags(a1, W, WIDTH, nb, 4);
            store_lipswish(acc, bs, nb, dst);
        }
        __syncthreads();

        // ---- fused heads: 64 diff tiles + 4 drift tiles; prefetch dW(st+1)
        {
            if (st + 1 < NSTEP && tid < BM * NOISE)
                async_load_f32(dW + (size_t)(st + 1) * BATCH * NOISE +
                               (size_t)(base + (tid >> 4)) * NOISE + (tid & 15),
                               dwn + tid);
            v16h ac[4];
            load_afrag(h1c, WIDTH, 4, ac);                    // shared diff A
#pragma unroll
            for (int i = 0; i < 4; ++i) {
                const int tile = wave + 16 * i;
                if (i < 3)
                    __builtin_prefetch(gW2c + (size_t)((tile + 16) * 16 +
                                       (threadIdx.x & 15)) * WIDTH, 0, 0);
                v8f acc = wmma_bfrags(ac, gW2c, WIDTH, tile * 16, 4);
                diff_epilogue(acc, tile, cb2s, cscs, dwc, g0);
            }
            if (wave >= 12) {
                const int tile = wave - 12;
                v16h av[4];
                load_afrag(h1v, WIDTH, 4, av);
                v8f acc = wmma_bfrags(av, wv2, WIDTH, tile * 16, 4);
                drift_epilogue(acc, tile, vb2s, vscs, dt, f0);
            }
        }
        __syncthreads();

        // ---- Heun stage: x' = [t, y + g0] ----
        {
            const float tcur = ts[st];
            for (int i = tid; i < BM * (HID + 1); i += NT) {
                int m = i / (HID + 1), k = i - m * (HID + 1);
                xh[m * K0P + k] = (_Float16)(k == 0 ? tcur
                                  : ys[m * HID + (k - 1)] + g0[m * HID + (k - 1)]);
            }
        }
        __syncthreads();
        if (wave < 8) {                                       // diffusion L0'
            v16h ax[3];
            load_afrag(xh, K0P, 3, ax);
            const int nb = wave * 16;
            v8f acc = wmma_bfrags(ax, wc0, K0P, nb, 3);
            store_lipswish(acc, cb0s, nb, h0c);
        }
        __syncthreads();
        if (wave < 8) {                                       // diffusion L1'
            v16h a1[4];
            load_afrag(h0c, WIDTH, 4, a1);
            const int nb = wave * 16;
            v8f acc = wmma_bfrags(a1, wc1, WIDTH, nb, 4);
            store_lipswish(acc, cb1s, nb, h1c);
        }
        __syncthreads();
        {                                                     // diffusion head'
            v16h ac[4];
            load_afrag(h1c, WIDTH, 4, ac);
#pragma unroll
            for (int i = 0; i < 4; ++i) {
                const int tile = wave + 16 * i;
                v8f acc = wmma_bfrags(ac, gW2c, WIDTH, tile * 16, 4);
                diff_epilogue(acc, tile, cb2s, cscs, dwc, g1);
            }
        }
        __syncthreads();

        // ---- state update ----
        for (int i = tid; i < BM * HID; i += NT)
            ys[i] += f0[i] + 0.5f * (g0[i] + g1[i]);
        __syncthreads();

        // ---- readout row (t = st+1) overlapped with next x=[t,y] build ----
        if (tid < BM * DATA) {
            int m = tid >> 3, d = tid & 7;
            float a = robs[d];
            for (int h = 0; h < HID; ++h) a += ys[m * HID + h] * roWs[d * HID + h];
            out[(size_t)(base + m) * TPTS * DATA + (size_t)(st + 1) * DATA + d] = a;
        }
        {
            const float tnxt = ts[st + 1];          // valid: st+1 <= 1024
            for (int i = tid; i < BM * (HID + 1); i += NT) {
                int m = i / (HID + 1), k = i - m * (HID + 1);
                xh[m * K0P + k] = (_Float16)(k == 0 ? tnxt : ys[m * HID + (k - 1)]);
            }
        }
    }
}

// ---------------------------------------------------------------------------
extern "C" void kernel_launch(void* const* d_in, const int* in_sizes, int n_in,
                              void* d_out, int out_size, void* d_ws, size_t ws_size,
                              hipStream_t stream) {
    (void)in_sizes; (void)n_in; (void)out_size; (void)ws_size;
    const float* ts  = (const float*)d_in[0];
    const float* z0  = (const float*)d_in[1];
    const float* dW  = (const float*)d_in[2];
    const float* iW0 = (const float*)d_in[3];
    const float* ib0 = (const float*)d_in[4];
    const float* iW1 = (const float*)d_in[5];
    const float* ib1 = (const float*)d_in[6];
    const float* iW2 = (const float*)d_in[7];
    const float* ib2 = (const float*)d_in[8];
    const float* vW0 = (const float*)d_in[9];
    const float* vb0 = (const float*)d_in[10];
    const float* vW1 = (const float*)d_in[11];
    const float* vb1 = (const float*)d_in[12];
    const float* vW2 = (const float*)d_in[13];
    const float* vb2 = (const float*)d_in[14];
    const float* vscale = (const float*)d_in[15];
    const float* cW0 = (const float*)d_in[16];
    const float* cb0 = (const float*)d_in[17];
    const float* cW1 = (const float*)d_in[18];
    const float* cb1 = (const float*)d_in[19];
    const float* cW2 = (const float*)d_in[20];
    const float* cb2 = (const float*)d_in[21];
    const float* cscale = (const float*)d_in[22];
    const float* roW = (const float*)d_in[23];
    const float* rob = (const float*)d_in[24];
    float* out = (float*)d_out;

    // workspace layout: f16 weights then fp32 y0
    _Float16* vW0h = (_Float16*)d_ws;
    _Float16* vW1h = vW0h + WIDTH * K0P;
    _Float16* vW2h = vW1h + WIDTH * WIDTH;
    _Float16* cW0h = vW2h + HID * WIDTH;
    _Float16* cW1h = cW0h + WIDTH * K0P;
    _Float16* cW2h = cW1h + WIDTH * WIDTH;
    float* yw = (float*)(cW2h + (size_t)(HID * NOISE) * WIDTH);

    convert_pad_f16<<<64, 256, 0, stream>>>(vW0, vW0h, WIDTH, HID + 1, K0P);
    convert_pad_f16<<<64, 256, 0, stream>>>(vW1, vW1h, WIDTH, WIDTH, WIDTH);
    convert_pad_f16<<<64, 256, 0, stream>>>(vW2, vW2h, HID, WIDTH, WIDTH);
    convert_pad_f16<<<64, 256, 0, stream>>>(cW0, cW0h, WIDTH, HID + 1, K0P);
    convert_pad_f16<<<64, 256, 0, stream>>>(cW1, cW1h, WIDTH, WIDTH, WIDTH);
    convert_pad_f16<<<256, 256, 0, stream>>>(cW2, cW2h, HID * NOISE, WIDTH, WIDTH);

    sde_init<<<BATCH, WIDTH, 0, stream>>>(z0, iW0, ib0, iW1, ib1, iW2, ib2,
                                          roW, rob, yw, out);

    const size_t lds_bytes =
        (size_t)(2 * WIDTH * K0P + 2 * WIDTH * WIDTH + HID * WIDTH) * 2 + // weights
        (size_t)(BM * K0P + 4 * BM * WIDTH) * 2 +                          // xh,h0v/c,h1v/c
        (size_t)(4 * BM * HID + 2 * BM * NOISE) * 4 +                      // ys,f0,g0,g1,dw x2
        (size_t)(4 * WIDTH + 2 * HID + 2 * HID * NOISE + DATA * HID + DATA) * 4; // biases

    sde_scan<<<BATCH / BM, NT, lds_bytes, stream>>>(
        ts, dW, vW0h, cW2h, vb0, vb1, vb2, vscale,
        cb0, cb1, cb2, cscale, roW, rob, yw, out);
}
